// StageBank_34084860461125
// MI455X (gfx1250) — matrix operand order repaired and checked by reading, not compiled, not run on hardware
//
#include <hip/hip_runtime.h>
#include <math.h>

// ---------------------------------------------------------------------------
// StageBank (7-stage top-2 routed MLP bank) for MI455X / gfx1250.
//   weighted = sum_s pi[:,s] * ( silu(h @ W1[s] + b1[s]) @ W2[s] + b2[s] )
//   evidence = weighted @ We + be
// Dense compute = 481 GFLOP, bytes ~190MB -> compute bound => bf16 WMMA path
// (v_wmma_f32_16x16x32_bf16, f32 accumulate). Per-stage weights fit in the
// 192MB L2, so weight re-reads across token tiles are L2-resident.
// Round 3: M=32 token tile -> each streamed B fragment feeds 2 WMMAs
// (2 A-row tiles), doubling FLOP per L2 byte (16 -> 32 FLOP/B) and doubling
// wmma density per loop body. ~194KB LDS (1 WG/WGP), 4-way N blocking kept.
// ---------------------------------------------------------------------------

typedef __attribute__((ext_vector_type(16))) __bf16 v16bf;
typedef __attribute__((ext_vector_type(8)))  float  v8f;

constexpr int kS   = 7;
constexpr int kB   = 4;
constexpr int kN   = 2048;
constexpr int kD   = 1024;
constexpr int kFF  = 2048;
constexpr int kTOK = kB * kN;          // 8192 tokens
constexpr int kM   = 32;               // tokens per workgroup (2 row tiles)

// LDS row strides (halfs), padded to de-phase bank access
constexpr int HS1 = kD  + 16;          // h tile rows   (32 x 1024)
constexpr int HS2 = kFF + 16;          // hid tile rows (32 x 2048)

__device__ __forceinline__ __bf16 f2bf(float f) {
  // round-to-nearest-even f32 -> bf16
  unsigned u = __builtin_bit_cast(unsigned, f);
  unsigned r = u + 0x7fffu + ((u >> 16) & 1u);
  return __builtin_bit_cast(__bf16, (unsigned short)(r >> 16));
}

// Build a 16x32 bf16 A fragment from an LDS tile (row-major, `stride` halfs).
// CDNA5 16-bit A layout: lane -> row M = lane&15; slot j ->
//   K = (j&7) + (j>=8 ? 16 : 0) + (lane>=16 ? 8 : 0)
// Per lane this is two contiguous 16B chunks -> compiler emits 2x ds_load_b128.
__device__ __forceinline__ v16bf load_a_frag(const __bf16* __restrict__ base,
                                             int stride, int kk, int lane,
                                             int row0) {
  const int half = lane >> 4;
  const int row  = row0 + (lane & 15);
  const __bf16* p = base + row * stride + kk * 32 + half * 8;
  v16bf a;
#pragma unroll
  for (int jj = 0; jj < 8; ++jj) {          // pair jj covers slots 2jj, 2jj+1
    int k = (jj < 4) ? (2 * jj) : (2 * jj + 8);
    a[2 * jj]     = p[k];
    a[2 * jj + 1] = p[k + 1];
  }
  return a;
}

// ---------------------------------------------------------------------------
// Pre-pack f32 weights (S,K,Ncols) into bf16 fragment-major B layout:
//   out[((s*(K/32)+kk)*(Ncols/16)+nt)*512 + lane*16 + j]
// CDNA5 16-bit B layout: lane = column (n=lane&15), slot j -> K = (j&7)+(j>=8?16:0)+8*(lane>=16)
// => each B fragment is one 32B contiguous load per lane (2x global_load_b128),
//    and the 4 adjacent column tiles used by register blocking are contiguous.
// ---------------------------------------------------------------------------
__global__ void pack_b_frags(const float* __restrict__ W, __bf16* __restrict__ out,
                             int K, int Ncols) {
  size_t tid = (size_t)blockIdx.x * blockDim.x + threadIdx.x;
  size_t total = (size_t)kS * K * Ncols;
  if (tid >= total) return;
  int    j    = (int)(tid & 15);
  int    lane = (int)((tid >> 4) & 31);
  size_t frag = tid >> 9;
  int ntc = Ncols / 16;
  int kkc = K / 32;
  int    nt = (int)(frag % ntc);
  size_t f2 = frag / ntc;
  int    kk = (int)(f2 % kkc);
  int    s  = (int)(f2 / kkc);
  int half = lane >> 4;
  int n    = lane & 15;
  int k = kk * 32 + (j & 7) + (j >> 3) * 16 + half * 8;
  out[tid] = f2bf(W[((size_t)s * K + k) * Ncols + nt * 16 + n]);
}

// ---------------------------------------------------------------------------
// Fused per-tile kernel: 32 tokens per workgroup, 8 waves, 7-stage loop.
// ---------------------------------------------------------------------------
__global__ void __launch_bounds__(256)
stagebank_fused(const float* __restrict__ h,  const float* __restrict__ pi,
                const __bf16* __restrict__ W1p, const float* __restrict__ b1,
                const __bf16* __restrict__ W2p, const float* __restrict__ b2,
                float* __restrict__ outw) {
  extern __shared__ __bf16 smem[];
  __bf16* sh_h   = smem;                  // 32 x kD  bf16, stride HS1 (~65KB)
  __bf16* sh_hid = smem + kM * HS1;       // 32 x kFF bf16, stride HS2 (~129KB)

  const int tid  = threadIdx.x;
  const int wid  = tid >> 5;              // wave id 0..7
  const int lane = tid & 31;
  const int half = lane >> 4;
  const int n    = lane & 15;
  const int tok0 = blockIdx.x * kM;

  // Stage h tile into LDS as bf16 once (reused by every stage).
  for (int idx = tid; idx < kM * kD; idx += 256) {
    int r = idx >> 10;                    // kD == 1024
    int c = idx & (kD - 1);
    sh_h[r * HS1 + c] = f2bf(h[(size_t)(tok0 + r) * kD + c]);
  }
  __syncthreads();

  v8f zero = {};
  v8f wacc0[8], wacc1[8];                 // weighted acc: 8 D-tiles x 2 row tiles
#pragma unroll
  for (int t = 0; t < 8; ++t) { wacc0[t] = zero; wacc1[t] = zero; }

  for (int s = 0; s < kS; ++s) {
    // ---- Phase 1: hid = silu(h @ W1[s] + b1[s])  (32 x 2048 into LDS) ----
    // wave owns FF tiles [wid*16, wid*16+15]; 4 passes of 4 blocked tiles,
    // each B fragment reused by both 16-row A tiles.
#pragma unroll 1
    for (int u = 0; u < 4; ++u) {
      const int nt0 = wid * 16 + u * 4;
      v8f acc0[4] = {zero, zero, zero, zero};
      v8f acc1[4] = {zero, zero, zero, zero};
      const __bf16* bp =
          W1p + (((size_t)s * 32) * 128 + nt0) * 512 + lane * 16;
#pragma unroll 2
      for (int kk = 0; kk < 32; ++kk) {
        v16bf a0 = load_a_frag(sh_h, HS1, kk, lane, 0);
        v16bf a1 = load_a_frag(sh_h, HS1, kk, lane, 16);
        const __bf16* bk = bp + (size_t)kk * (128 * 512);
        __builtin_prefetch(bk + 128 * 512, 0, 1);
#pragma unroll
        for (int c = 0; c < 4; ++c) {
          v16bf b = *(const v16bf*)(bk + c * 512);
          acc0[c] = __builtin_amdgcn_wmma_f32_16x16x32_bf16(
              false, a0, false, b, (short)0, acc0[c], false, false);
          acc1[c] = __builtin_amdgcn_wmma_f32_16x16x32_bf16(
              false, a1, false, b, (short)0, acc1[c], false, false);
        }
      }
#pragma unroll
      for (int c = 0; c < 4; ++c) {
        const int   nt = nt0 + c;
        const float bb = b1[s * kFF + nt * 16 + n];
#pragma unroll
        for (int i = 0; i < 8; ++i) {     // rows m and m+16, col = nt*16+n
          int   m  = i + half * 8;
          float x0 = acc0[c][i] + bb;
          float x1 = acc1[c][i] + bb;
          sh_hid[m * HS2 + nt * 16 + n]        = f2bf(x0 / (1.f + __expf(-x0)));
          sh_hid[(m + 16) * HS2 + nt * 16 + n] = f2bf(x1 / (1.f + __expf(-x1)));
        }
      }
    }
    __syncthreads();

    // routing weights for this stage (per output row of the C/D layout)
    float pis0[8], pis1[8];
#pragma unroll
    for (int i = 0; i < 8; ++i) {
      int m = i + half * 8;
      pis0[i] = pi[(size_t)(tok0 + m) * kS + s];
      pis1[i] = pi[(size_t)(tok0 + m + 16) * kS + s];
    }

    // ---- Phase 2: wacc += pi_s * (hid @ W2[s] + b2[s]) ----
    // wave owns D tiles [wid*8, wid*8+7]; 2 passes of 4 blocked tiles x 2 rows.
#pragma unroll 1
    for (int u = 0; u < 2; ++u) {
      const int dt0 = wid * 8 + u * 4;
      v8f acc0[4] = {zero, zero, zero, zero};
      v8f acc1[4] = {zero, zero, zero, zero};
      const __bf16* bp =
          W2p + (((size_t)s * 64) * 64 + dt0) * 512 + lane * 16;
#pragma unroll 2
      for (int kk = 0; kk < 64; ++kk) {
        v16bf a0 = load_a_frag(sh_hid, HS2, kk, lane, 0);
        v16bf a1 = load_a_frag(sh_hid, HS2, kk, lane, 16);
        const __bf16* bk = bp + (size_t)kk * (64 * 512);
        __builtin_prefetch(bk + 64 * 512, 0, 1);
#pragma unroll
        for (int c = 0; c < 4; ++c) {
          v16bf b = *(const v16bf*)(bk + c * 512);
          acc0[c] = __builtin_amdgcn_wmma_f32_16x16x32_bf16(
              false, a0, false, b, (short)0, acc0[c], false, false);
          acc1[c] = __builtin_amdgcn_wmma_f32_16x16x32_bf16(
              false, a1, false, b, (short)0, acc1[c], false, false);
        }
      }
#pragma unroll
      for (int c = 0; c < 4; ++c) {
        const int   dt  = dt0 + c;
        const float bb2 = b2[s * kD + dt * 16 + n];
#pragma unroll
        for (int i = 0; i < 8; ++i) {
          wacc0[u * 4 + c][i] += pis0[i] * (acc0[c][i] + bb2);
          wacc1[u * 4 + c][i] += pis1[i] * (acc1[c][i] + bb2);
        }
      }
    }
    __syncthreads();                      // protect sh_hid before next stage
  }

  // write weighted (f32), both row tiles
#pragma unroll
  for (int t = 0; t < 8; ++t) {
    const int dt = wid * 8 + t;
#pragma unroll
    for (int i = 0; i < 8; ++i) {
      int m = i + half * 8;
      outw[(size_t)(tok0 + m) * kD + dt * 16 + n]      = wacc0[t][i];
      outw[(size_t)(tok0 + m + 16) * kD + dt * 16 + n] = wacc1[t][i];
    }
  }
}

// ---------------------------------------------------------------------------
// evidence[tok,s] = dot(weighted[tok,:], We[:,s]) + be[s]   (tiny: 117 MFLOP)
// ---------------------------------------------------------------------------
__global__ void evidence_kernel(const float* __restrict__ w,
                                const float* __restrict__ We,
                                const float* __restrict__ be,
                                float* __restrict__ ev) {
  int t = blockIdx.x * blockDim.x + threadIdx.x;
  if (t >= kTOK * kS) return;
  int tok = t / kS, s = t % kS;
  const float* wr = w + (size_t)tok * kD;
  float sum = be[s];
#pragma unroll 4
  for (int d = 0; d < kD; ++d) sum += wr[d] * We[d * kS + s];
  ev[t] = sum;
}

extern "C" void kernel_launch(void* const* d_in, const int* in_sizes, int n_in,
                              void* d_out, int out_size, void* d_ws, size_t ws_size,
                              hipStream_t stream) {
  const float* h  = (const float*)d_in[0];   // (B,N,D)
  const float* pi = (const float*)d_in[1];   // (B,N,7)
  const float* W1 = (const float*)d_in[2];   // (7,D,FF)
  const float* b1 = (const float*)d_in[3];   // (7,FF)
  const float* W2 = (const float*)d_in[4];   // (7,FF,D)
  const float* b2 = (const float*)d_in[5];   // (7,D)
  const float* We = (const float*)d_in[6];   // (D,7)
  const float* be = (const float*)d_in[7];   // (7,)

  float* outw = (float*)d_out;                       // (B,N,D) weighted
  float* ev   = outw + (size_t)kTOK * kD;            // (B,N,7) evidence

  __bf16* W1p = (__bf16*)d_ws;                       // 29.36 MB
  __bf16* W2p = W1p + (size_t)kS * kD * kFF;         // 29.36 MB (58.7 MB total)

  const size_t wn = (size_t)kS * kD * kFF;           // elements per weight bank
  pack_b_frags<<<(unsigned)((wn + 255) / 256), 256, 0, stream>>>(W1, W1p, kD, kFF);
  pack_b_frags<<<(unsigned)((wn + 255) / 256), 256, 0, stream>>>(W2, W2p, kFF, kD);

  const int smem_bytes = (kM * HS1 + kM * HS2) * (int)sizeof(__bf16); // ~194 KB
  stagebank_fused<<<kTOK / kM, 256, smem_bytes, stream>>>(
      h, pi, W1p, b1, W2p, b2, outw);

  evidence_kernel<<<(kTOK * kS + 255) / 256, 256, 0, stream>>>(outw, We, be, ev);
}
